// TemporalAttentionLayer_59760174957094
// MI455X (gfx1250) — compile-verified, matrix operand order checked by reading, not run on hardware
//
#include <hip/hip_runtime.h>

// ---------------------------------------------------------------------------
// TemporalAttentionLayer for MI455X (gfx1250): bf16 WMMA pipeline + TDM.
// B=2, T=8, C=512, H=W=16 -> tokens M = B*T*H*W = 4096, heads=8, head_dim=64.
// ---------------------------------------------------------------------------

typedef __attribute__((ext_vector_type(16))) __bf16 v16bf;
typedef __attribute__((ext_vector_type(8)))  float  v8f;
typedef __attribute__((ext_vector_type(4)))  unsigned int v4u;
typedef __attribute__((ext_vector_type(8)))  int    v8i_t;
typedef __attribute__((ext_vector_type(4)))  int    v4i_t;

#define DEV __device__ __forceinline__

static constexpr int Bc   = 2;
static constexpr int Tc   = 8;
static constexpr int Cc   = 512;
static constexpr int Lc   = 256;          // H*W
static constexpr int Ntok = Tc * Lc;      // 2048 tokens per batch
static constexpr int Mrow = Bc * Ntok;    // 4096 rows total
static constexpr int HEADS = 8;
static constexpr int HD   = 64;

DEV v8f v8f_zero() {
    v8f z;
#pragma unroll
    for (int i = 0; i < 8; ++i) z[i] = 0.0f;
    return z;
}

union Frag16 { v16bf v; uint4 u[2]; };

// ---------------------------------------------------------------------------
// Tensor Data Mover: 2D tile (elem=2B) global -> LDS, optional 4-DWORD pad
// after every 32 DWORDs (row stride 64 bf16 -> 72 bf16 in LDS).
// D# layout per CDNA5 ISA 8.3/8.4 (group0 128b, group1 256b).
// Toolchain builtin is the 6-arg clang-23 form:
//   (uint32x4 g0, int32x8 g1, int32x4 g2, int32x4 g3, int32x8 g4, i32 cpol)
// ---------------------------------------------------------------------------
#if __has_builtin(__builtin_amdgcn_tensor_load_to_lds)
#define HAVE_TDM 1
#else
#define HAVE_TDM 0
#endif

DEV void tdm_load_tile_2d(unsigned int lds_byte_addr, const void* global_tile,
                          unsigned int tile_d0, unsigned int tile_d1,
                          unsigned int tensor_d0, unsigned int tensor_d1,
                          unsigned int stride_elems, bool pad72) {
#if HAVE_TDM
    unsigned long long ga = (unsigned long long)(uintptr_t)global_tile;
    v4u g0;
    g0[0] = 1u;                                        // count=1, is_restore=0
    g0[1] = lds_byte_addr;                             // LDS dest byte address
    g0[2] = (unsigned int)(ga & 0xFFFFFFFFu);          // global_addr[31:0]
    g0[3] = (unsigned int)((ga >> 32) & 0x1FFFFFFu)    // global_addr[56:32]
            | (2u << 30);                              // type = 2 ("image")
    unsigned int w0 = (1u << 16);                      // data_size=1 (2 bytes)
    if (pad72) w0 |= (1u << 20)                        // pad_enable
                   | (4u << 22)                        // pad_interval: 32 DWORDs
                   | (3u << 25);                       // pad_amount: 4 DWORDs
    v8i_t g1;
    g1[0] = (int)w0;
    g1[1] = (int)((tensor_d0 & 0xFFFFu) << 16);                       // dim0[15:0]
    g1[2] = (int)((tensor_d0 >> 16) | ((tensor_d1 & 0xFFFFu) << 16)); // dim0 hi / dim1 lo
    g1[3] = (int)((tensor_d1 >> 16) | (tile_d0 << 16));               // dim1 hi / tile0
    g1[4] = (int)(tile_d1 & 0xFFFFu);                                 // tile1, tile2=0
    g1[5] = (int)stride_elems;                                        // dim0_stride[31:0]
    g1[6] = 0;
    g1[7] = 0;
    v4i_t z4;
    z4[0] = 0; z4[1] = 0; z4[2] = 0; z4[3] = 0;
    v8i_t z8;
#pragma unroll
    for (int i = 0; i < 8; ++i) z8[i] = 0;
    __builtin_amdgcn_tensor_load_to_lds(g0, g1, z4, z4, z8, 0);
#endif
}

DEV void tdm_wait() {
#if HAVE_TDM
    __builtin_amdgcn_s_wait_tensorcnt(0);
#endif
}

DEV unsigned int lds_addr_of(const void* p) {
    return (unsigned int)(uintptr_t)p;   // low 32 bits of generic addr = LDS offset
}

// ---------------------------------------------------------------------------
// Weight convert + transpose: W[K,N] f32 -> Wt[N,K] bf16  (coalesced writes)
// ---------------------------------------------------------------------------
__global__ void wtrans_kernel(const float* __restrict__ W, __bf16* __restrict__ Wt,
                              int Kdim, int Ndim) {
    int idx = blockIdx.x * 256 + threadIdx.x;
    if (idx >= Kdim * Ndim) return;
    int j = idx / Kdim;
    int k = idx - j * Kdim;
    Wt[idx] = (__bf16)W[(size_t)k * Ndim + j];
}

// ---------------------------------------------------------------------------
// Permute [B,T,C,H,W] f32 -> token-major x[M, C] f32 (coalesced reads)
// ---------------------------------------------------------------------------
__global__ void permute_in_kernel(const float* __restrict__ f, float* __restrict__ x) {
    int idx = blockIdx.x * 256 + threadIdx.x;   // < 2*8*512*256 = 2^21
    int l = idx & 255;
    int c = (idx >> 8) & 511;
    int t = (idx >> 17) & 7;
    int b = idx >> 20;
    int row = b * Ntok + t * Lc + l;
    x[(size_t)row * Cc + c] = f[idx];
}

// token-major y[M, C] f32 -> [B,T,C,H,W] f32 (coalesced writes)
__global__ void permute_out_kernel(const float* __restrict__ y, float* __restrict__ out) {
    int idx = blockIdx.x * 256 + threadIdx.x;
    int l = idx & 255;
    int c = (idx >> 8) & 511;
    int t = (idx >> 17) & 7;
    int b = idx >> 20;
    int row = b * Ntok + t * Lc + l;
    out[idx] = y[(size_t)row * Cc + c];
}

// ---------------------------------------------------------------------------
// LayerNorm over C=512, one block (256 thr) per row, output bf16.
// ---------------------------------------------------------------------------
__global__ __launch_bounds__(256)
void ln_kernel(const float* __restrict__ X, const float* __restrict__ g,
               const float* __restrict__ be, __bf16* __restrict__ Y) {
    const int row = blockIdx.x;
    const int tid = threadIdx.x;
    const float* xr = X + (size_t)row * Cc;
    float v0 = xr[tid];
    float v1 = xr[tid + 256];

    __shared__ float red[256];
    red[tid] = v0 + v1;
    __syncthreads();
#pragma unroll
    for (int s = 128; s > 0; s >>= 1) {
        if (tid < s) red[tid] += red[tid + s];
        __syncthreads();
    }
    float mu = red[0] * (1.0f / (float)Cc);
    __syncthreads();

    float d0 = v0 - mu, d1 = v1 - mu;
    red[tid] = d0 * d0 + d1 * d1;
    __syncthreads();
#pragma unroll
    for (int s = 128; s > 0; s >>= 1) {
        if (tid < s) red[tid] += red[tid + s];
        __syncthreads();
    }
    float rs = rsqrtf(red[0] * (1.0f / (float)Cc) + 1e-5f);

    __bf16* yr = Y + (size_t)row * Cc;
    yr[tid]       = (__bf16)(d0 * rs * g[tid]       + be[tid]);
    yr[tid + 256] = (__bf16)(d1 * rs * g[tid + 256] + be[tid + 256]);
}

// ---------------------------------------------------------------------------
// Tiled WMMA GEMM:  out = epilogue( A[M,K]bf16 @ Wt[N,K]^T bf16 + bias )
// BM=128, BN=128, BK=32; 256 threads = 8 waves; wave -> 32x64 patch:
// 2 A-frags reused across 4 B-frags -> 8 WMMAs per K-step per wave.
// EPI 0: scatter into Q/K/V bf16 [B*H, N, d]
// EPI 1: GELU(exact) -> bf16 row-major
// EPI 2: residual f32 add -> f32 row-major
// ---------------------------------------------------------------------------
template <int EPI>
__global__ __launch_bounds__(256)
void gemm_kernel(const __bf16* __restrict__ A, const __bf16* __restrict__ Wt,
                 const float* __restrict__ bias, int Kdim, int Ncols,
                 float* __restrict__ out_f32, const float* __restrict__ res_f32,
                 __bf16* __restrict__ out_bf16,
                 __bf16* __restrict__ qd, __bf16* __restrict__ kd, __bf16* __restrict__ vd) {
    constexpr int BM = 128, BN = 128, BK = 32, LDA = BK + 8;
    __shared__ __bf16 lA[BM * LDA];
    __shared__ __bf16 lB[BN * LDA];

    const int tid  = threadIdx.x;
    const int m0   = blockIdx.y * BM;
    const int n0   = blockIdx.x * BN;
    const int wid  = tid >> 5;
    const int lane = tid & 31;
    const int wr   = wid & 3;       // 4 row groups of 32 rows
    const int wc   = wid >> 2;      // 2 col groups of 64 cols
    const int colL = lane & 15;
    const int hi   = lane >> 4;

    v8f acc[2][4];
#pragma unroll
    for (int i = 0; i < 2; ++i)
#pragma unroll
        for (int j = 0; j < 4; ++j) acc[i][j] = v8f_zero();

    const int srow = tid >> 1, sseg = (tid & 1) * 16;    // 128 rows x 32 cols staging
    const __bf16* Ap = A  + (size_t)(m0 + srow) * Kdim + sseg;
    const __bf16* Bp = Wt + (size_t)(n0 + srow) * Kdim + sseg;

    const int nK = Kdim / BK;
    for (int kb = 0; kb < nK; ++kb) {
        const uint4* ga = (const uint4*)(Ap + kb * BK);
        const uint4* gb = (const uint4*)(Bp + kb * BK);
        uint4 a0 = ga[0], a1 = ga[1];
        uint4 b0 = gb[0], b1 = gb[1];
        *(uint4*)&lA[srow * LDA + sseg]     = a0;
        *(uint4*)&lA[srow * LDA + sseg + 8] = a1;
        *(uint4*)&lB[srow * LDA + sseg]     = b0;
        *(uint4*)&lB[srow * LDA + sseg + 8] = b1;
        if (kb + 1 < nK) {
            __builtin_prefetch(Ap + (size_t)(kb + 1) * BK, 0, 0);
            __builtin_prefetch(Bp + (size_t)(kb + 1) * BK, 0, 0);
        }
        __syncthreads();

        Frag16 fa[2], fb[4];
#pragma unroll
        for (int i = 0; i < 2; ++i) {
            int off = (wr * 32 + i * 16 + colL) * LDA + hi * 16;
            fa[i].u[0] = *(const uint4*)&lA[off];
            fa[i].u[1] = *(const uint4*)&lA[off + 8];
        }
#pragma unroll
        for (int j = 0; j < 4; ++j) {
            int off = (wc * 64 + j * 16 + colL) * LDA + hi * 16;
            fb[j].u[0] = *(const uint4*)&lB[off];
            fb[j].u[1] = *(const uint4*)&lB[off + 8];
        }
#pragma unroll
        for (int i = 0; i < 2; ++i)
#pragma unroll
            for (int j = 0; j < 4; ++j)
                acc[i][j] = __builtin_amdgcn_wmma_f32_16x16x32_bf16(
                    false, fa[i].v, false, fb[j].v, (short)0, acc[i][j], false, false);
        __syncthreads();
    }

    // epilogue — C layout: VGPR r: lanes 0-15 row r, lanes 16-31 row r+8; col = lane&15
#pragma unroll
    for (int i = 0; i < 2; ++i)
#pragma unroll
        for (int j = 0; j < 4; ++j) {
            const int rbase = m0 + wr * 32 + i * 16 + hi * 8;
            const int gc    = n0 + wc * 64 + j * 16 + colL;
            const float bs  = bias[gc];
#pragma unroll
            for (int r = 0; r < 8; ++r) {
                const int gr = rbase + r;
                float val = acc[i][j][r] + bs;
                if (EPI == 0) {
                    int bb = gr >> 11, n = gr & 2047;
                    int sel = gc >> 9, jj = gc & 511;
                    int head = jj >> 6, dd = jj & 63;
                    __bf16* dst = (sel == 0) ? qd : (sel == 1) ? kd : vd;
                    dst[((size_t)(bb * HEADS + head) * Ntok + n) * HD + dd] = (__bf16)val;
                } else if (EPI == 1) {
                    float gl = 0.5f * val * (1.0f + erff(val * 0.70710678118f));
                    out_bf16[(size_t)gr * Ncols + gc] = (__bf16)gl;
                } else {
                    out_f32[(size_t)gr * Ncols + gc] =
                        res_f32[(size_t)gr * Ncols + gc] + val;
                }
            }
        }
}

// ---------------------------------------------------------------------------
// Flash attention: one block (4 waves) per (bh, 64-query tile).
// Q/K tiles staged by the Tensor Data Mover (hardware-padded to stride 72);
// V staged manually with transpose; online softmax on WMMA C layout;
// P restaged through per-wave LDS strip as A-fragments; O += P @ V.
// ---------------------------------------------------------------------------
__global__ __launch_bounds__(128)
void attn_kernel(const __bf16* __restrict__ Qg, const __bf16* __restrict__ Kg,
                 const __bf16* __restrict__ Vg, __bf16* __restrict__ Og) {
    constexpr int LDS = 72;  // 64 + 8 pad
    __shared__ __bf16 lQ[64 * LDS];
    __shared__ __bf16 lK[64 * LDS];
    __shared__ __bf16 lVt[64 * LDS];   // transposed: [d][key]
    __shared__ __bf16 lP[64 * LDS];    // per-wave 16-row strips

    const int qt   = blockIdx.x;       // 0..31
    const int bh   = blockIdx.y;       // 0..15
    const int b    = bh >> 3, h = bh & 7;
    const int tid  = threadIdx.x;
    const int wave = tid >> 5;
    const int lane = tid & 31;
    const int colL = lane & 15;
    const int hi   = lane >> 4;

    const __bf16* Qb = Qg + (size_t)bh * Ntok * HD;
    const __bf16* Kb = Kg + (size_t)bh * Ntok * HD;
    const __bf16* Vb = Vg + (size_t)bh * Ntok * HD;

    // stage Q tile (64 x 64) via TDM (wave 0 issues; TENSORcnt fenced)
    if (HAVE_TDM) {
        if (wave == 0) {
            tdm_load_tile_2d(lds_addr_of(&lQ[0]), Qb + (size_t)qt * 64 * HD,
                             64, 64, HD, Ntok, HD, true);
            tdm_wait();
        }
    } else {
        int row = tid >> 1, cb = (tid & 1) * 32;
        const uint4* s = (const uint4*)(Qb + (size_t)(qt * 64 + row) * HD + cb);
        uint4* d = (uint4*)&lQ[row * LDS + cb];
        d[0] = s[0]; d[1] = s[1]; d[2] = s[2]; d[3] = s[3];
    }
    __syncthreads();

    Frag16 fq[2];
#pragma unroll
    for (int kk = 0; kk < 2; ++kk) {
        int off = (wave * 16 + colL) * LDS + kk * 32 + hi * 16;
        fq[kk].u[0] = *(const uint4*)&lQ[off];
        fq[kk].u[1] = *(const uint4*)&lQ[off + 8];
    }

    v8f o0 = v8f_zero(), o1 = v8f_zero(), o2 = v8f_zero(), o3 = v8f_zero();
    float m[8], l[8];
#pragma unroll
    for (int r = 0; r < 8; ++r) { m[r] = -1e30f; l[r] = 0.0f; }
    const float scale = 0.125f;   // 1/sqrt(64)

    for (int kt = 0; kt < Ntok / 64; ++kt) {
        // K tile via TDM; V tile manual (needs transpose)
        if (HAVE_TDM) {
            if (wave == 0)
                tdm_load_tile_2d(lds_addr_of(&lK[0]), Kb + (size_t)kt * 64 * HD,
                                 64, 64, HD, Ntok, HD, true);
        } else {
            int row = tid >> 1, cb = (tid & 1) * 32;
            const uint4* ks = (const uint4*)(Kb + (size_t)(kt * 64 + row) * HD + cb);
            uint4* kdst = (uint4*)&lK[row * LDS + cb];
            kdst[0] = ks[0]; kdst[1] = ks[1]; kdst[2] = ks[2]; kdst[3] = ks[3];
        }
        {
            int row = tid >> 1, cb = (tid & 1) * 32;
            union { uint4 u[4]; __bf16 hb[32]; } t;
            const uint4* vs = (const uint4*)(Vb + (size_t)(kt * 64 + row) * HD + cb);
            t.u[0] = vs[0]; t.u[1] = vs[1]; t.u[2] = vs[2]; t.u[3] = vs[3];
#pragma unroll
            for (int e = 0; e < 32; ++e) lVt[(cb + e) * LDS + row] = t.hb[e];
        }
        if (HAVE_TDM && wave == 0) tdm_wait();
        __syncthreads();

        // S strip = Q(16x64) @ K^T(64x64): 4 tiles x 2 K-chunks
        v8f s[4];
#pragma unroll
        for (int j = 0; j < 4; ++j) {
            s[j] = v8f_zero();
#pragma unroll
            for (int kk = 0; kk < 2; ++kk) {
                Frag16 fk;
                int off = (j * 16 + colL) * LDS + kk * 32 + hi * 16;
                fk.u[0] = *(const uint4*)&lK[off];
                fk.u[1] = *(const uint4*)&lK[off + 8];
                s[j] = __builtin_amdgcn_wmma_f32_16x16x32_bf16(
                    false, fq[kk].v, false, fk.v, (short)0, s[j], false, false);
            }
#pragma unroll
            for (int r = 0; r < 8; ++r) s[j][r] *= scale;
        }

        // online softmax (row stats across the 16-lane column group)
#pragma unroll
        for (int r = 0; r < 8; ++r) {
            float t = fmaxf(fmaxf(s[0][r], s[1][r]), fmaxf(s[2][r], s[3][r]));
            t = fmaxf(t, __shfl_xor(t, 1, 32));
            t = fmaxf(t, __shfl_xor(t, 2, 32));
            t = fmaxf(t, __shfl_xor(t, 4, 32));
            t = fmaxf(t, __shfl_xor(t, 8, 32));
            float mn = fmaxf(m[r], t);
            float alpha = __expf(m[r] - mn);
            m[r] = mn;
            float rs = 0.0f;
#pragma unroll
            for (int j = 0; j < 4; ++j) {
                float p = __expf(s[j][r] - mn);
                s[j][r] = p;
                rs += p;
            }
            rs += __shfl_xor(rs, 1, 32);
            rs += __shfl_xor(rs, 2, 32);
            rs += __shfl_xor(rs, 4, 32);
            rs += __shfl_xor(rs, 8, 32);
            l[r] = l[r] * alpha + rs;
            o0[r] *= alpha; o1[r] *= alpha; o2[r] *= alpha; o3[r] *= alpha;
        }

        // restage P (C layout -> A layout) through this wave's private LDS strip
#pragma unroll
        for (int j = 0; j < 4; ++j)
#pragma unroll
            for (int r = 0; r < 8; ++r)
                lP[(wave * 16 + hi * 8 + r) * LDS + j * 16 + colL] = (__bf16)s[j][r];

        Frag16 fp[2];
#pragma unroll
        for (int kk = 0; kk < 2; ++kk) {
            int off = (wave * 16 + colL) * LDS + kk * 32 + hi * 16;
            fp[kk].u[0] = *(const uint4*)&lP[off];
            fp[kk].u[1] = *(const uint4*)&lP[off + 8];
        }
#pragma unroll
        for (int n = 0; n < 4; ++n) {
            v8f* oo = (n == 0) ? &o0 : (n == 1) ? &o1 : (n == 2) ? &o2 : &o3;
#pragma unroll
            for (int kk = 0; kk < 2; ++kk) {
                Frag16 fv;
                int off = (n * 16 + colL) * LDS + kk * 32 + hi * 16;
                fv.u[0] = *(const uint4*)&lVt[off];
                fv.u[1] = *(const uint4*)&lVt[off + 8];
                *oo = __builtin_amdgcn_wmma_f32_16x16x32_bf16(
                    false, fp[kk].v, false, fv.v, (short)0, *oo, false, false);
            }
        }
        __syncthreads();
    }

    // normalize + store into token-major O[M, C] bf16 (col = h*64 + d)
#pragma unroll
    for (int n = 0; n < 4; ++n) {
        v8f ov = (n == 0) ? o0 : (n == 1) ? o1 : (n == 2) ? o2 : o3;
#pragma unroll
        for (int r = 0; r < 8; ++r) {
            int q = qt * 64 + wave * 16 + hi * 8 + r;
            float val = ov[r] / l[r];
            Og[((size_t)(b * Ntok + q)) * Cc + h * HD + n * 16 + colL] = (__bf16)val;
        }
    }
}

// ---------------------------------------------------------------------------
// Host launcher
// ---------------------------------------------------------------------------
extern "C" void kernel_launch(void* const* d_in, const int* in_sizes, int n_in,
                              void* d_out, int out_size, void* d_ws, size_t ws_size,
                              hipStream_t stream) {
    const float* features = (const float*)d_in[0];
    const float* qkv_w  = (const float*)d_in[1];
    const float* qkv_b  = (const float*)d_in[2];
    const float* proj_w = (const float*)d_in[3];
    const float* proj_b = (const float*)d_in[4];
    const float* ln1_g  = (const float*)d_in[5];
    const float* ln1_b  = (const float*)d_in[6];
    const float* w1     = (const float*)d_in[7];
    const float* b1     = (const float*)d_in[8];
    const float* w2     = (const float*)d_in[9];
    const float* b2     = (const float*)d_in[10];
    const float* ln2_g  = (const float*)d_in[11];
    const float* ln2_b  = (const float*)d_in[12];
    float* out = (float*)d_out;

    char* ws = (char*)d_ws;
    size_t off = 0;
    auto take = [&](size_t bytes) -> char* {
        char* p = ws + off;
        off += (bytes + 255) & ~size_t(255);
        return p;
    };

    float*  x_f32   = (float*) take((size_t)Mrow * Cc * 4);         // permuted input / residual 1
    __bf16* xn      = (__bf16*)take((size_t)Mrow * Cc * 2);         // LN1 output
    __bf16* qkvWt   = (__bf16*)take((size_t)3 * Cc * Cc * 2);       // [1536,512]
    __bf16* projWt  = (__bf16*)take((size_t)Cc * Cc * 2);           // [512,512]
    __bf16* w1t     = (__bf16*)take((size_t)4 * Cc * Cc * 2);       // [2048,512]
    __bf16* w2t     = (__bf16*)take((size_t)4 * Cc * Cc * 2);       // [512,2048]
    __bf16* Qb      = (__bf16*)take((size_t)Bc * HEADS * Ntok * HD * 2);
    __bf16* Kb      = (__bf16*)take((size_t)Bc * HEADS * Ntok * HD * 2);
    __bf16* Vb      = (__bf16*)take((size_t)Bc * HEADS * Ntok * HD * 2);
    __bf16* Ob      = (__bf16*)take((size_t)Mrow * Cc * 2);         // attn out
    float*  x2_f32  = (float*) take((size_t)Mrow * Cc * 4);         // residual 2
    __bf16* h2      = (__bf16*)take((size_t)Mrow * Cc * 2);         // LN2 output
    __bf16* mid     = (__bf16*)take((size_t)Mrow * 4 * Cc * 2);     // GELU(fc1)
    float*  y_f32   = (float*) take((size_t)Mrow * Cc * 4);         // final token-major
    (void)ws_size; (void)in_sizes; (void)n_in; (void)out_size;

    // 1) weights -> transposed bf16
    wtrans_kernel<<<(3 * Cc * Cc + 255) / 256, 256, 0, stream>>>(qkv_w,  qkvWt,  Cc,     3 * Cc);
    wtrans_kernel<<<(Cc * Cc     + 255) / 256, 256, 0, stream>>>(proj_w, projWt, Cc,     Cc);
    wtrans_kernel<<<(4 * Cc * Cc + 255) / 256, 256, 0, stream>>>(w1,     w1t,    Cc,     4 * Cc);
    wtrans_kernel<<<(4 * Cc * Cc + 255) / 256, 256, 0, stream>>>(w2,     w2t,    4 * Cc, Cc);

    // 2) permute input + LN1
    permute_in_kernel<<<(Mrow * Cc) / 256, 256, 0, stream>>>(features, x_f32);
    ln_kernel<<<Mrow, 256, 0, stream>>>(x_f32, ln1_g, ln1_b, xn);

    // 3) QKV GEMM (4096 x 512 x 1536) -> Q/K/V bf16
    gemm_kernel<0><<<dim3(3 * Cc / 128, Mrow / 128), 256, 0, stream>>>(
        xn, qkvWt, qkv_b, Cc, 3 * Cc, nullptr, nullptr, nullptr, Qb, Kb, Vb);

    // 4) attention
    attn_kernel<<<dim3(Ntok / 64, Bc * HEADS), 128, 0, stream>>>(Qb, Kb, Vb, Ob);

    // 5) proj GEMM + residual -> x2
    gemm_kernel<2><<<dim3(Cc / 128, Mrow / 128), 256, 0, stream>>>(
        Ob, projWt, proj_b, Cc, Cc, x2_f32, x_f32, nullptr, nullptr, nullptr, nullptr);

    // 6) LN2
    ln_kernel<<<Mrow, 256, 0, stream>>>(x2_f32, ln2_g, ln2_b, h2);

    // 7) fc1 + GELU -> mid (4096 x 512 x 2048)
    gemm_kernel<1><<<dim3(4 * Cc / 128, Mrow / 128), 256, 0, stream>>>(
        h2, w1t, b1, Cc, 4 * Cc, nullptr, nullptr, mid, nullptr, nullptr, nullptr);

    // 8) fc2 + residual -> y (4096 x 2048 x 512)
    gemm_kernel<2><<<dim3(Cc / 128, Mrow / 128), 256, 0, stream>>>(
        mid, w2t, b2, 4 * Cc, Cc, y_f32, x2_f32, nullptr, nullptr, nullptr, nullptr);

    // 9) permute back
    permute_out_kernel<<<(Mrow * Cc) / 256, 256, 0, stream>>>(y_f32, out);
}
